// GAT_37056977830623
// MI455X (gfx1250) — matrix operand order repaired and checked by reading, not compiled, not run on hardware
//
#include <hip/hip_runtime.h>

// ---------------------------------------------------------------------------
// GAT (2x GATConv + linear head) for MI455X / gfx1250.
//  - GEMMs: bf16 WMMA (v_wmma_f32_16x16x32_bf16), 128x64 block tile, 8 waves.
//    Weights pre-packed to bf16 in WMMA fragment-tile order; per k-step the
//    4 KB fragment block is moved with GLOBAL_LOAD_ASYNC_TO_LDS_B128 and
//    consumed via ds_load_b128. Activations pre-packed to bf16 row-major.
//  - Edge softmax: orderable-uint atomicMax + fp32 atomicAdd (L2-resident:
//    node feature arrays are 76.8 MB each, both fit in the 192 MB L2).
// ---------------------------------------------------------------------------

typedef __attribute__((ext_vector_type(16))) __bf16 v16bf;
typedef __attribute__((ext_vector_type(8)))  float  v8f;

constexpr int   N_NODES   = 50000;
constexpr int   N_EDGES   = 800000;
constexpr int   EP        = N_EDGES + N_NODES;   // + self loops
constexpr int   IN_DIM    = 256;
constexpr int   HID       = 128;
constexpr int   HEADS     = 3;
constexpr int   HC        = HEADS * HID;         // 384
constexpr int   NCLS      = 40;
constexpr float NEG_SLOPE = 0.2f;

// GEMM block tile
constexpr int BM = 128;    // 8 waves x 16 rows
constexpr int BN = 64;     // 4 WMMA n-tiles
constexpr int FRAG = 32 * 16;  // 512 bf16 elements per fragment tile (1 KB)

// fp32 -> bf16, round-to-nearest-even (bit manipulation: always compiles)
__device__ __forceinline__ __bf16 f2bf(float f) {
    unsigned u = __float_as_uint(f);
    u += 0x7fffu + ((u >> 16) & 1u);
    unsigned short hs = (unsigned short)(u >> 16);
    __bf16 r;
    __builtin_memcpy(&r, &hs, sizeof(r));
    return r;
}

// order-preserving float <-> uint mapping for atomicMax on floats
__device__ __forceinline__ unsigned fmap(float f) {
    unsigned u = __float_as_uint(f);
    return (u & 0x80000000u) ? ~u : (u | 0x80000000u);
}
__device__ __forceinline__ float funmap(unsigned m) {
    unsigned u = (m & 0x80000000u) ? (m & 0x7fffffffu) : ~m;
    return __uint_as_float(u);
}

// ---------------------------------------------------------------------------
// Pack fp32 -> bf16 row-major (activations). 8 elements per thread.
// total must be a multiple of 8 (50000*256 / 50000*384: yes).
// ---------------------------------------------------------------------------
__global__ void k_pack_a(const float* __restrict__ X, __bf16* __restrict__ Y,
                         int total)
{
    const int i = (blockIdx.x * blockDim.x + threadIdx.x) * 8;
    if (i >= total) return;
    float4 f0 = *(const float4*)(X + i);
    float4 f1 = *(const float4*)(X + i + 4);
    __bf16 o[8] = { f2bf(f0.x), f2bf(f0.y), f2bf(f0.z), f2bf(f0.w),
                    f2bf(f1.x), f2bf(f1.y), f2bf(f1.z), f2bf(f1.w) };
    int4 pk;
    __builtin_memcpy(&pk, o, 16);
    *(int4*)(Y + i) = pk;
}

// ---------------------------------------------------------------------------
// Pack W[K,N] fp32 -> bf16 in WMMA B-fragment-tile order:
//   Wbf[(kt*NTpad + nt)*512 + lane*16 + e]  where
//   k = kt*32 + (lane>=16 ? 16 : 0) + e,  n = nt*16 + (lane & 15).
// Zero-fills n >= N (column padding for the N=40 head).
// ---------------------------------------------------------------------------
__global__ void k_pack_w(const float* __restrict__ W, __bf16* __restrict__ Wbf,
                         int K, int N, int NTpad)
{
    const int Ktiles = K >> 5;
    const int total  = Ktiles * NTpad * FRAG;
    const int t = blockIdx.x * blockDim.x + threadIdx.x;
    if (t >= total) return;
    const int frag = t >> 9;           // /512
    const int r    = t & 511;
    const int lane = r >> 4;
    const int e    = r & 15;
    const int kt   = frag / NTpad;
    const int nt   = frag - kt * NTpad;
    const int k    = (kt << 5) + ((lane >> 4) << 4) + e;
    const int n    = (nt << 4) + (lane & 15);
    const int nc   = (n < N) ? n : (N - 1);
    float w = W[(size_t)k * N + nc];
    w = (n < N) ? w : 0.0f;
    Wbf[t] = f2bf(w);
}

// ---------------------------------------------------------------------------
// C[M,N] = Abf[M,K](bf16) @ Wbf(frag-tiled bf16) (+ bias[N]), f32 accumulate.
// Block: 256 threads = 8 waves; block tile 128M x 64N.
// Per k-step: 4 KB of pre-swizzled B fragments moved global->LDS with
// GLOBAL_LOAD_ASYNC_TO_LDS_B128 (16 B per thread), A fragment = 2x b128
// global loads, then 4x (2x ds_load_b128 + v_wmma_f32_16x16x32_bf16).
// ---------------------------------------------------------------------------
__global__ __launch_bounds__(256)
void k_gemm_bf16wmma(const __bf16* __restrict__ Abf,
                     const __bf16* __restrict__ Wbf,
                     const float* __restrict__ bias, float* __restrict__ C,
                     int M, int K, int N, int NTpad)
{
    __shared__ __bf16 Bs[4 * FRAG];      // 4 fragments = 4 KB

    const int tid  = threadIdx.x;
    const int wave = tid >> 5;
    const int lane = tid & 31;
    const int half = lane >> 4;          // 0: lanes 0-15, 1: lanes 16-31
    const int l16  = lane & 15;

    const int tm  = blockIdx.x * BM + wave * 16;
    const int nt0 = blockIdx.y * 4;      // first of 4 fragment columns
    const int tn0 = nt0 * 16;

    const int     mrow  = tm + l16;
    const int     arow  = (mrow < M) ? mrow : (M - 1);  // clamp; store guarded
    const __bf16* ArowB = Abf + (size_t)arow * K;

    // LDS byte offset of this thread's 16 B async-copy destination
    // (low 32 bits of the flat address are the LDS offset).
    const unsigned ldsDst = (unsigned)(uintptr_t)(&Bs[0]) + (unsigned)tid * 16u;

    const int Ktiles = K >> 5;
    v8f acc0 = {}, acc1 = {}, acc2 = {}, acc3 = {};

    for (int kt = 0; kt < Ktiles; ++kt) {
        // ---- async copy: 4 contiguous fragment tiles (4 KB), 16 B per thread
        const __bf16* gsrc =
            Wbf + (((size_t)(kt * NTpad + nt0)) << 9) + (size_t)tid * 8;
        asm volatile("global_load_async_to_lds_b128 %0, %1, off"
                     :: "v"(ldsDst), "v"(gsrc) : "memory");

        // ---- A fragment (bf16, contiguous k order as in k_pack_w): two 16 B chunks per lane.
        // lanes 0-15: K = k0+[0..15]; lanes 16-31: K = k0+[16..31].
        const int ab = (kt << 5) + half * 16;
        int4 c0 = *(const int4*)(ArowB + ab);
        int4 c1 = *(const int4*)(ArowB + ab + 8);
        v16bf a;
        __builtin_memcpy(&a, &c0, 16);
        __builtin_memcpy(reinterpret_cast<char*>(&a) + 16, &c1, 16);

        asm volatile("s_wait_asynccnt 0x0" ::: "memory");
        __syncthreads();

        // ---- B fragments from LDS (32 B/lane each -> 2x ds_load_b128)
        const v16bf* bp = (const v16bf*)Bs;
        const v16bf b0 = bp[0 * 32 + lane];
        const v16bf b1 = bp[1 * 32 + lane];
        const v16bf b2 = bp[2 * 32 + lane];
        const v16bf b3 = bp[3 * 32 + lane];

        acc0 = __builtin_amdgcn_wmma_f32_16x16x32_bf16(false, a, false, b0,
                                                       (short)0, acc0, false, false);
        acc1 = __builtin_amdgcn_wmma_f32_16x16x32_bf16(false, a, false, b1,
                                                       (short)0, acc1, false, false);
        acc2 = __builtin_amdgcn_wmma_f32_16x16x32_bf16(false, a, false, b2,
                                                       (short)0, acc2, false, false);
        acc3 = __builtin_amdgcn_wmma_f32_16x16x32_bf16(false, a, false, b3,
                                                       (short)0, acc3, false, false);
        __syncthreads();   // LDS consumed; safe to overwrite next k-step
    }

    // ---- store: C/D layout VGPR v -> row tm + v + half*8, col = tile*16 + l16
    v8f accs[4] = {acc0, acc1, acc2, acc3};
#pragma unroll
    for (int t = 0; t < 4; ++t) {
        const int col = tn0 + t * 16 + l16;
        if (col < N) {
            const float bv = bias ? bias[col] : 0.0f;
#pragma unroll
            for (int v = 0; v < 8; ++v) {
                const int row = tm + v + half * 8;
                if (row < M) C[(size_t)row * N + col] = accs[t][v] + bv;
            }
        }
    }
}

// ---------------------------------------------------------------------------
// a_src[n,h] / a_dst[n,h] dot products. One wave per (node, head):
// coalesced 32-wide loads + shfl_xor tree reduction (wave32).
// ---------------------------------------------------------------------------
__global__ __launch_bounds__(256)
void k_dots(const float* __restrict__ h, const float* __restrict__ atts,
            const float* __restrict__ attd, float* __restrict__ asrc,
            float* __restrict__ adst)
{
    const int gw   = (blockIdx.x * blockDim.x + threadIdx.x) >> 5;  // wave id
    const int lane = threadIdx.x & 31;
    if (gw >= N_NODES * HEADS) return;
    const int n  = gw / HEADS;
    const int hh = gw - n * HEADS;

    const float* hp = h + (size_t)n * HC + hh * HID;
    const float* sp = atts + hh * HID;
    const float* dp = attd + hh * HID;

    float s = 0.f, d = 0.f;
#pragma unroll
    for (int c = lane; c < HID; c += 32) {
        const float v = hp[c];
        s += v * sp[c];
        d += v * dp[c];
    }
#pragma unroll
    for (int o = 16; o > 0; o >>= 1) {
        s += __shfl_xor(s, o, 32);
        d += __shfl_xor(d, o, 32);
    }
    if (lane == 0) { asrc[gw] = s; adst[gw] = d; }
}

// out[n, j] = bias[j]  (accumulator init)
__global__ void k_bias_init(float* __restrict__ o, const float* __restrict__ bias)
{
    const int t = blockIdx.x * blockDim.x + threadIdx.x;
    if (t >= N_NODES * HC) return;
    o[t] = bias[t % HC];
}

__device__ __forceinline__ void edge_sd(int e, const long long* src,
                                        const long long* dst, int& s, int& d)
{
    if (e < N_EDGES) { s = (int)src[e]; d = (int)dst[e]; }
    else             { s = e - N_EDGES; d = s; }          // self loop
}

// segment max of leaky_relu(a_src[s]+a_dst[d]) over dst
__global__ void k_edge_max(const long long* __restrict__ src,
                           const long long* __restrict__ dst,
                           const float* __restrict__ asrc,
                           const float* __restrict__ adst,
                           unsigned* __restrict__ amaxU)
{
    const int t = blockIdx.x * blockDim.x + threadIdx.x;
    if (t >= EP * HEADS) return;
    const int e = t / HEADS, hh = t - e * HEADS;
    int s, d; edge_sd(e, src, dst, s, d);
    float a = asrc[s * HEADS + hh] + adst[d * HEADS + hh];
    a = (a > 0.f) ? a : NEG_SLOPE * a;
    atomicMax(&amaxU[d * HEADS + hh], fmap(a));
}

// ex = exp(alpha - amax[dst]); denom[dst] += ex
__global__ void k_edge_exp(const long long* __restrict__ src,
                           const long long* __restrict__ dst,
                           const float* __restrict__ asrc,
                           const float* __restrict__ adst,
                           const unsigned* __restrict__ amaxU,
                           float* __restrict__ ex, float* __restrict__ denom)
{
    const int t = blockIdx.x * blockDim.x + threadIdx.x;
    if (t >= EP * HEADS) return;
    const int e = t / HEADS, hh = t - e * HEADS;
    int s, d; edge_sd(e, src, dst, s, d);
    float a = asrc[s * HEADS + hh] + adst[d * HEADS + hh];
    a = (a > 0.f) ? a : NEG_SLOPE * a;
    const float ev = __expf(a - funmap(amaxU[d * HEADS + hh]));
    ex[t] = ev;
    atomicAdd(&denom[d * HEADS + hh], ev);
}

// hout[dst, h, c] += hlin[src, h, c] * (ex / (denom[dst,h] + 1e-16))
// one block (384 threads = 12 waves) per edge; thread t -> (h = t/128, c = t%128)
__global__ __launch_bounds__(HC)
void k_edge_accum(const long long* __restrict__ src,
                  const long long* __restrict__ dst,
                  const float* __restrict__ ex, const float* __restrict__ denom,
                  const float* __restrict__ hlin, float* __restrict__ hout)
{
    const int e  = blockIdx.x;
    const int t  = threadIdx.x;       // 0..383
    const int hh = t >> 7;            // head
    int s, d; edge_sd(e, src, dst, s, d);
    const float attn = ex[(size_t)e * HEADS + hh] /
                       (denom[d * HEADS + hh] + 1e-16f);
    const float v = hlin[(size_t)s * HC + t] * attn;
    atomicAdd(&hout[(size_t)d * HC + t], v);
}

__global__ void k_relu(float* __restrict__ p, int n)
{
    const int t = blockIdx.x * blockDim.x + threadIdx.x;
    if (t >= n) return;
    const float v = p[t];
    p[t] = (v > 0.f) ? v : 0.f;
}

// ---------------------------------------------------------------------------
// Host-side drivers (all async on `stream`; graph-capture safe).
// ---------------------------------------------------------------------------
static void run_gemm(const float* X, const float* W, const float* bias,
                     float* C, __bf16* Abf, __bf16* Wbf,
                     int M, int K, int N, int NTpad, hipStream_t stream)
{
    const int atot = M * K;
    k_pack_a<<<(atot / 8 + 255) / 256, 256, 0, stream>>>(X, Abf, atot);
    const int wtot = (K >> 5) * NTpad * FRAG;
    k_pack_w<<<(wtot + 255) / 256, 256, 0, stream>>>(W, Wbf, K, N, NTpad);
    dim3 gg((M + BM - 1) / BM, NTpad / 4);
    k_gemm_bf16wmma<<<gg, 256, 0, stream>>>(Abf, Wbf, bias, C, M, K, N, NTpad);
}

static void gat_layer(const float* X, int K, const float* W, const float* atts,
                      const float* attd, const float* bias,
                      const long long* srcE, const long long* dstE,
                      float* hlin, float* hout, float* asrc, float* adst,
                      unsigned* amaxU, float* denom, float* ex,
                      __bf16* Abf, __bf16* Wbf, hipStream_t stream)
{
    run_gemm(X, W, nullptr, hlin, Abf, Wbf, N_NODES, K, HC, HC / 16, stream);

    const int nwaves = N_NODES * HEADS;               // one wave each
    k_dots<<<(nwaves + 7) / 8, 256, 0, stream>>>(hlin, atts, attd, asrc, adst);

    hipMemsetAsync(amaxU, 0, sizeof(unsigned) * (size_t)N_NODES * HEADS, stream);
    hipMemsetAsync(denom, 0, sizeof(float)    * (size_t)N_NODES * HEADS, stream);
    k_bias_init<<<(N_NODES * HC + 255) / 256, 256, 0, stream>>>(hout, bias);

    const int et = EP * HEADS;
    k_edge_max<<<(et + 255) / 256, 256, 0, stream>>>(srcE, dstE, asrc, adst, amaxU);
    k_edge_exp<<<(et + 255) / 256, 256, 0, stream>>>(srcE, dstE, asrc, adst, amaxU, ex, denom);
    k_edge_accum<<<EP, HC, 0, stream>>>(srcE, dstE, ex, denom, hlin, hout);
    k_relu<<<(N_NODES * HC + 255) / 256, 256, 0, stream>>>(hout, N_NODES * HC);
}

extern "C" void kernel_launch(void* const* d_in, const int* in_sizes, int n_in,
                              void* d_out, int out_size, void* d_ws, size_t ws_size,
                              hipStream_t stream)
{
    (void)in_sizes; (void)n_in; (void)out_size; (void)ws_size;

    const float*     x    = (const float*)d_in[0];
    const long long* ei   = (const long long*)d_in[1];   // int64 [2, E]
    const float*     W1   = (const float*)d_in[2];
    const float*     as1  = (const float*)d_in[3];
    const float*     ad1  = (const float*)d_in[4];
    const float*     b1   = (const float*)d_in[5];
    const float*     W2   = (const float*)d_in[6];
    const float*     as2  = (const float*)d_in[7];
    const float*     ad2  = (const float*)d_in[8];
    const float*     b2   = (const float*)d_in[9];
    const float*     outW = (const float*)d_in[10];
    const float*     outb = (const float*)d_in[11];
    float*           out  = (float*)d_out;

    const long long* srcE = ei;
    const long long* dstE = ei + N_EDGES;

    // workspace carving (~205 MB)
    char* ws = (char*)d_ws;
    float*    hlin  = (float*)ws;    ws += sizeof(float)    * (size_t)N_NODES * HC;
    float*    hout  = (float*)ws;    ws += sizeof(float)    * (size_t)N_NODES * HC;
    float*    asrc  = (float*)ws;    ws += sizeof(float)    * (size_t)N_NODES * HEADS;
    float*    adst  = (float*)ws;    ws += sizeof(float)    * (size_t)N_NODES * HEADS;
    unsigned* amaxU = (unsigned*)ws; ws += sizeof(unsigned) * (size_t)N_NODES * HEADS;
    float*    denom = (float*)ws;    ws += sizeof(float)    * (size_t)N_NODES * HEADS;
    float*    ex    = (float*)ws;    ws += sizeof(float)    * (size_t)EP * HEADS;
    __bf16*   Abf   = (__bf16*)ws;   ws += sizeof(__bf16)   * (size_t)N_NODES * HC;
    __bf16*   Wbf   = (__bf16*)ws;   ws += sizeof(__bf16)   * (size_t)(HC / 32) * (HC / 16) * FRAG;

    // layer 1: x -> hout (relu'd)
    gat_layer(x, IN_DIM, W1, as1, ad1, b1, srcE, dstE,
              hlin, hout, asrc, adst, amaxU, denom, ex, Abf, Wbf, stream);
    // layer 2: hout -> hlin -> hout (relu'd)
    gat_layer(hout, HC, W2, as2, ad2, b2, srcE, dstE,
              hlin, hout, asrc, adst, amaxU, denom, ex, Abf, Wbf, stream);

    // output projection: out = relu(hout) @ outW + outb   (hout already relu'd)
    // N = 40 padded to 4 fragment columns (64)
    run_gemm(hout, outW, outb, out, Abf, Wbf, N_NODES, HC, NCLS, 4, stream);
}
